// RegBase_50964081935487
// MI455X (gfx1250) — matrix operand order repaired and checked by reading, not compiled, not run on hardware
//
#include <hip/hip_runtime.h>

#define DIM      128
#define MTILE    128
#define SH_PITCH 136   // bf16 elements per row; 272B row pitch keeps 16B alignment

#if defined(__AMDGCN__) && __has_builtin(__builtin_amdgcn_global_load_async_to_lds_b64) && \
    __has_builtin(__builtin_amdgcn_s_wait_asynccnt)
#define HAS_ASYNC_LDS 1
#else
#define HAS_ASYNC_LDS 0
#endif

typedef __attribute__((ext_vector_type(16))) __bf16 v16bf;
typedef __attribute__((ext_vector_type(8)))  float  v8f;
typedef __attribute__((ext_vector_type(4)))  float  v4f;
typedef __attribute__((ext_vector_type(2)))  int    v2i;

// Native conversion: fptrunc f32 -> bf16 lowers to v_cvt_pk_bf16_f32 on gfx1250.
__device__ __forceinline__ __bf16 f2bf(float f) { return (__bf16)f; }

__device__ __forceinline__ float silu(float x) { return x / (1.0f + __expf(-x)); }

#if HAS_ASYNC_LDS
// ASYNCcnt-tracked 8-byte DMA: global -> LDS (b64 moves a <2 x i32>).
__device__ __forceinline__ void asyncLoadB64(const long long* g, long long* l)
{
    __builtin_amdgcn_global_load_async_to_lds_b64(
        (__attribute__((address_space(1))) v2i*)(g),
        (__attribute__((address_space(3))) v2i*)(l), 0, 0);
}
#endif

union AFrag {
    v16bf  v;
    uint4  q[2];
    __bf16 e[16];
};

__device__ __forceinline__ v16bf loadB(const __bf16* __restrict__ wf, int fragIdx, int lane)
{
    return *(const v16bf*)(wf + ((size_t)fragIdx * 32 + lane) * 16);
}

// Build a 16x32 bf16 A-fragment from a row of f32 (non-temporal streaming read).
__device__ __forceinline__ void buildA(AFrag& a, const float* __restrict__ r, int k0)
{
    const v4f p0 = __builtin_nontemporal_load((const v4f*)(r + k0));
    const v4f p1 = __builtin_nontemporal_load((const v4f*)(r + k0 + 4));
    const v4f p2 = __builtin_nontemporal_load((const v4f*)(r + k0 + 16));
    const v4f p3 = __builtin_nontemporal_load((const v4f*)(r + k0 + 20));
    a.e[0]  = f2bf(p0[0]); a.e[1]  = f2bf(p0[1]); a.e[2]  = f2bf(p0[2]); a.e[3]  = f2bf(p0[3]);
    a.e[4]  = f2bf(p1[0]); a.e[5]  = f2bf(p1[1]); a.e[6]  = f2bf(p1[2]); a.e[7]  = f2bf(p1[3]);
    a.e[8]  = f2bf(p2[0]); a.e[9]  = f2bf(p2[1]); a.e[10] = f2bf(p2[2]); a.e[11] = f2bf(p2[3]);
    a.e[12] = f2bf(p3[0]); a.e[13] = f2bf(p3[1]); a.e[14] = f2bf(p3[2]); a.e[15] = f2bf(p3[3]);
}

// ---------------------------------------------------------------------------
// Pre-swizzle W (DIM x DIM row-major f32) into per-lane WMMA B-fragments:
// frag[(nt*4+kt)*32 + lane][e] = bf16( W[(kt*32 + (lane>>4)*8 + e + (e&8))*DIM + nt*16 + (lane&15)] )
// ---------------------------------------------------------------------------
__global__ __launch_bounds__(256) void prep_kernel(
    const float* __restrict__ W1, const float* __restrict__ W2,
    const float* __restrict__ W3,
    __bf16* __restrict__ f1, __bf16* __restrict__ f2, __bf16* __restrict__ f3)
{
    const float* W = (blockIdx.x == 0) ? W1 : (blockIdx.x == 1) ? W2 : W3;
    __bf16*      F = (blockIdx.x == 0) ? f1 : (blockIdx.x == 1) ? f2 : f3;

    for (int idx = threadIdx.x; idx < 8 * 4 * 32; idx += blockDim.x) {
        const int lane = idx & 31;
        const int kt   = (idx >> 5) & 3;
        const int nt   = idx >> 7;
        const int n     = nt * 16 + (lane & 15);
        const int kbase = kt * 32 + (lane >> 4) * 8;
        #pragma unroll
        for (int e = 0; e < 16; e++)
            F[(size_t)idx * 16 + e] = f2bf(W[(size_t)(kbase + e + (e & 8)) * DIM + n]);
    }
}

__global__ __launch_bounds__(256) void zero_kernel(float4* __restrict__ p, int n4)
{
    int i = blockIdx.x * blockDim.x + threadIdx.x;
    if (i < n4) p[i] = make_float4(0.f, 0.f, 0.f, 0.f);
}

// ---------------------------------------------------------------------------
// Per-atom MLP: h = silu(feat@W1+b1); h2 = h@W2+b2; pooled[batch[i]] += h2
// One block = 128 atom rows, 8 waves x 16 rows, full 128-dim width.
// ---------------------------------------------------------------------------
__global__ __launch_bounds__(256) void atom_mlp_kernel(
    const float* __restrict__ feat, const long long* __restrict__ batch,
    const float* __restrict__ b1,   const float* __restrict__ b2,
    const __bf16* __restrict__ w1f, const __bf16* __restrict__ w2f,
    float* __restrict__ pooled, int nAtoms)
{
    __shared__ __bf16    sH[MTILE * SH_PITCH];   // silu(layer1) tile, bf16
    __shared__ long long sB[MTILE];              // batch ids for this tile

    const int t    = threadIdx.x;
    const int wave = t >> 5;
    const int lane = t & 31;
    const int blockRow = blockIdx.x * MTILE;

    if (t < MTILE) {
        int gi = blockRow + t;
        if (gi > nAtoms - 1) gi = nAtoms - 1;
#if HAS_ASYNC_LDS
        asyncLoadB64(batch + gi, &sB[t]);
#else
        sB[t] = batch[gi];
#endif
    }

    // ----- A-fragment addressing (16x32 bf16 layout) -----
    const int aRow  = lane & 15;
    const int kHalf = lane >> 4;
    const int rowBase = blockRow + wave * 16;
    int gRow = rowBase + aRow;
    if (gRow > nAtoms - 1) gRow = nAtoms - 1;
    const float* fr = feat + (size_t)gRow * DIM;

    const v8f vzero = {0.f, 0.f, 0.f, 0.f, 0.f, 0.f, 0.f, 0.f};
    v8f acc[8];
    #pragma unroll
    for (int nt = 0; nt < 8; nt++) acc[nt] = vzero;

    // ----- layer 1: feat @ W1 (double-buffered B fragments) -----
    v16bf bcur = loadB(w1f, 0, lane);
    #pragma unroll
    for (int kt = 0; kt < 4; kt++) {
        AFrag a;
        buildA(a, fr, kt * 32 + kHalf * 8);
        #pragma unroll
        for (int nt = 0; nt < 8; nt++) {
            const int nidx = (nt < 7) ? ((nt + 1) * 4 + kt) : ((kt < 3) ? (kt + 1) : 0);
            v16bf bnext = loadB(w1f, nidx, lane);
            acc[nt] = __builtin_amdgcn_wmma_f32_16x16x32_bf16(
                false, a.v, false, bcur, (short)0, acc[nt], false, false);
            bcur = bnext;
        }
    }

    // ----- bias + SiLU, stash as bf16 in LDS (C-layout scatter) -----
    const int cN   = lane & 15;
    const int mOff = (lane >> 4) * 8;
    #pragma unroll
    for (int nt = 0; nt < 8; nt++) {
        const int n = nt * 16 + cN;
        const float bias = b1[n];
        #pragma unroll
        for (int j = 0; j < 8; j++)
            sH[(wave * 16 + mOff + j) * SH_PITCH + n] = f2bf(silu(acc[nt][j] + bias));
    }
#if HAS_ASYNC_LDS
    __builtin_amdgcn_s_wait_asynccnt(0);   // batch-id DMA landed before barrier
#endif
    __syncthreads();

    // ----- layer 2: h @ W2 (A from LDS via 16B-aligned b128 reads) -----
    #pragma unroll
    for (int nt = 0; nt < 8; nt++) acc[nt] = vzero;

    const int hBase = (wave * 16 + aRow) * SH_PITCH;
    bcur = loadB(w2f, 0, lane);
    #pragma unroll
    for (int kt = 0; kt < 4; kt++) {
        AFrag a;
        const int k0 = kt * 32 + kHalf * 8;
        a.q[0] = *(const uint4*)(&sH[hBase + k0]);
        a.q[1] = *(const uint4*)(&sH[hBase + k0 + 16]);
        #pragma unroll
        for (int nt = 0; nt < 8; nt++) {
            const int nidx = (nt < 7) ? ((nt + 1) * 4 + kt) : ((kt < 3) ? (kt + 1) : 0);
            v16bf bnext = loadB(w2f, nidx, lane);
            acc[nt] = __builtin_amdgcn_wmma_f32_16x16x32_bf16(
                false, a.v, false, bcur, (short)0, acc[nt], false, false);
            bcur = bnext;
        }
    }

    // ----- bias + segment sum; batch is sorted, so run-length-merge equal
    //       segment ids within each lane's 8 consecutive rows before the atomic.
    long long gseg[8];
    #pragma unroll
    for (int j = 0; j < 8; j++) gseg[j] = sB[wave * 16 + mOff + j];

    #pragma unroll
    for (int nt = 0; nt < 8; nt++) {
        const int n = nt * 16 + cN;
        const float bias = b2[n];
        long long cur = gseg[0];
        float run = 0.f;
        #pragma unroll
        for (int j = 0; j < 8; j++) {
            const int row = rowBase + mOff + j;
            const float v = (row < nAtoms) ? (acc[nt][j] + bias) : 0.f;
            if (gseg[j] != cur) {
                atomicAdd(pooled + (size_t)cur * DIM + n, run);
                cur = gseg[j];
                run = 0.f;
            }
            run += v;
        }
        atomicAdd(pooled + (size_t)cur * DIM + n, run);
    }
}

// ---------------------------------------------------------------------------
// Graph head: out = silu(pooled@W3 + b3) @ W4 + b4   (W4 is [128,1])
// One block = 128 graph rows; WMMA for the 128x128 GEMM, LDS-atomic row dot.
// ---------------------------------------------------------------------------
__global__ __launch_bounds__(256) void head_kernel(
    const float* __restrict__ pooled, const __bf16* __restrict__ w3f,
    const float* __restrict__ b3, const float* __restrict__ W4,
    const float* __restrict__ b4, float* __restrict__ out, int nGraphs)
{
    __shared__ float sOut[MTILE];

    const int t    = threadIdx.x;
    const int wave = t >> 5;
    const int lane = t & 31;

    if (t < MTILE) sOut[t] = 0.f;
    __syncthreads();

    const int aRow  = lane & 15;
    const int kHalf = lane >> 4;
    const int rowBase = blockIdx.x * MTILE + wave * 16;
    int gRow = rowBase + aRow;
    if (gRow > nGraphs - 1) gRow = nGraphs - 1;
    const float* pr = pooled + (size_t)gRow * DIM;

    const v8f vzero = {0.f, 0.f, 0.f, 0.f, 0.f, 0.f, 0.f, 0.f};
    v8f acc[8];
    #pragma unroll
    for (int nt = 0; nt < 8; nt++) acc[nt] = vzero;

    v16bf bcur = loadB(w3f, 0, lane);
    #pragma unroll
    for (int kt = 0; kt < 4; kt++) {
        AFrag a;
        buildA(a, pr, kt * 32 + kHalf * 8);
        #pragma unroll
        for (int nt = 0; nt < 8; nt++) {
            const int nidx = (nt < 7) ? ((nt + 1) * 4 + kt) : ((kt < 3) ? (kt + 1) : 0);
            v16bf bnext = loadB(w3f, nidx, lane);
            acc[nt] = __builtin_amdgcn_wmma_f32_16x16x32_bf16(
                false, a.v, false, bcur, (short)0, acc[nt], false, false);
            bcur = bnext;
        }
    }

    // silu(+b3), scale by W4[n], reduce over n into per-row scalars via LDS atomics
    const int cN   = lane & 15;
    const int mOff = (lane >> 4) * 8;
    float part[8] = {0.f, 0.f, 0.f, 0.f, 0.f, 0.f, 0.f, 0.f};
    #pragma unroll
    for (int nt = 0; nt < 8; nt++) {
        const int n = nt * 16 + cN;
        const float bias = b3[n];
        const float w4   = W4[n];
        #pragma unroll
        for (int j = 0; j < 8; j++)
            part[j] += silu(acc[nt][j] + bias) * w4;
    }
    #pragma unroll
    for (int j = 0; j < 8; j++)
        atomicAdd(&sOut[wave * 16 + mOff + j], part[j]);
    __syncthreads();

    if (t < MTILE) {
        const int r = blockIdx.x * MTILE + t;
        if (r < nGraphs) out[r] = sOut[t] + b4[0];
    }
}

// ---------------------------------------------------------------------------
extern "C" void kernel_launch(void* const* d_in, const int* in_sizes, int n_in,
                              void* d_out, int out_size, void* d_ws, size_t ws_size,
                              hipStream_t stream)
{
    const float*     feat  = (const float*)d_in[0];
    const long long* batch = (const long long*)d_in[1];   // reference dtype: int64
    const float* W1 = (const float*)d_in[2]; const float* b1 = (const float*)d_in[3];
    const float* W2 = (const float*)d_in[4]; const float* b2 = (const float*)d_in[5];
    const float* W3 = (const float*)d_in[6]; const float* b3 = (const float*)d_in[7];
    const float* W4 = (const float*)d_in[8]; const float* b4 = (const float*)d_in[9];

    const int nAtoms  = in_sizes[0] / DIM;
    const int nGraphs = out_size - nAtoms * DIM;   // tuple: [nGraphs] out + [nAtoms*DIM] feat

    // workspace layout: pooled f32 accumulator, then 3x 32KB weight-fragment buffers
    char*  ws     = (char*)d_ws;
    float* pooled = (float*)ws;
    size_t off = (((size_t)nGraphs * DIM * sizeof(float)) + 255) & ~(size_t)255;
    const size_t FRAG_BYTES = (size_t)8 * 4 * 32 * 16 * sizeof(__bf16);   // 32 KB
    __bf16* w1f = (__bf16*)(ws + off); off += FRAG_BYTES;
    __bf16* w2f = (__bf16*)(ws + off); off += FRAG_BYTES;
    __bf16* w3f = (__bf16*)(ws + off); off += FRAG_BYTES;
    (void)ws_size; (void)n_in;

    prep_kernel<<<3, 256, 0, stream>>>(W1, W2, W3, w1f, w2f, w3f);

    const int n4 = (nGraphs * DIM) / 4;
    zero_kernel<<<(n4 + 255) / 256, 256, 0, stream>>>((float4*)pooled, n4);

    atom_mlp_kernel<<<(nAtoms + MTILE - 1) / MTILE, 256, 0, stream>>>(
        feat, batch, b1, b2, w1f, w2f, pooled, nAtoms);

    head_kernel<<<(nGraphs + MTILE - 1) / MTILE, 256, 0, stream>>>(
        pooled, w3f, b3, W4, b4, (float*)d_out, nGraphs);

    // second tuple output: identity passthrough of feat
    (void)hipMemcpyAsync((float*)d_out + nGraphs, feat,
                         (size_t)nAtoms * DIM * sizeof(float),
                         hipMemcpyDeviceToDevice, stream);
}
